// SelfAttention_v1_63084479643816
// MI455X (gfx1250) — matrix operand order repaired and checked by reading, not compiled, not run on hardware
//
#include <hip/hip_runtime.h>

typedef __attribute__((ext_vector_type(16))) _Float16 v16h;
typedef __attribute__((ext_vector_type(8)))  _Float16 v8h;
typedef __attribute__((ext_vector_type(8)))  float    v8f;
typedef __attribute__((ext_vector_type(4)))  int      v4i;
typedef __attribute__((ext_vector_type(4)))  unsigned u32x4;
typedef __attribute__((ext_vector_type(8)))  int      i32x8;
typedef __attribute__((ext_vector_type(4)))  int      i32x4;

#define SEQ_N 4096
#define DIM   768

// ---------------------------------------------------------------------------
// gfx1250 async global->LDS copy (16B per lane), with sync fallback
// ---------------------------------------------------------------------------
#if defined(__has_builtin)
#if __has_builtin(__builtin_amdgcn_global_load_async_to_lds_b128)
#define HAVE_ASYNC_LDS 1
#endif
#if __has_builtin(__builtin_amdgcn_tensor_load_to_lds)
#define HAVE_TDM 1
#endif
#endif

__device__ __forceinline__ void copy16_to_lds(_Float16* lds, const _Float16* g) {
#if defined(HAVE_ASYNC_LDS)
  __builtin_amdgcn_global_load_async_to_lds_b128((v4i*)g, (v4i*)lds, 0, 0);
#else
  *(v8h*)lds = *(const v8h*)g;
#endif
}

__device__ __forceinline__ void wait_async_lds() {
#if defined(HAVE_ASYNC_LDS)
#if __has_builtin(__builtin_amdgcn_s_wait_asynccnt)
  __builtin_amdgcn_s_wait_asynccnt(0);
#else
  asm volatile("s_wait_asynccnt 0" ::: "memory");
#endif
#endif
}

// ---------------------------------------------------------------------------
// TDM: 2D tile load, f16 elements, LDS rows padded 64B payload + 16B pad.
// D# packing per CDNA5 ISA 8.3/8.4 (group0 128b, group1 256b, groups2/3 = 0).
// ---------------------------------------------------------------------------
__device__ __forceinline__ void tdm_load_tile_f16(unsigned lds_byte_off,
                                                  const _Float16* gaddr,
                                                  unsigned tile_d0,   // elems/row
                                                  unsigned tile_d1,   // rows
                                                  unsigned long long stride_d0) {
#if defined(HAVE_TDM)
  unsigned long long ga = (unsigned long long)(uintptr_t)gaddr;
  u32x4 g0;
  g0.x = 1u;                                            // count=1 user D#
  g0.y = lds_byte_off;                                  // lds_addr
  g0.z = (unsigned)ga;                                  // global_addr[31:0]
  g0.w = (unsigned)((ga >> 32) & 0x01FFFFFFu) | 0x80000000u;  // [56:32] | type=2
  i32x8 g1;
  // data_size=1 (2B) | pad_enable | pad_interval=3 (16 dw = 64B) | pad_amount=3 (4 dw = 16B)
  g1[0] = (int)((1u << 16) | (1u << 20) | (3u << 22) | (3u << 25));
  g1[1] = (int)((tile_d0 & 0xFFFFu) << 16);             // tensor_dim0[15:0]
  g1[2] = (int)((tile_d0 >> 16) | ((tile_d1 & 0xFFFFu) << 16));  // dim0 hi | dim1 lo
  g1[3] = (int)((tile_d1 >> 16) | (tile_d0 << 16));     // dim1 hi | tile_dim0
  g1[4] = (int)tile_d1;                                 // tile_dim1 | tile_dim2=0
  g1[5] = (int)(unsigned)stride_d0;                     // dim0_stride[31:0]
  g1[6] = (int)(unsigned)((stride_d0 >> 32) & 0xFFFFu); // dim0_stride[47:32] | dim1_stride lo=0
  g1[7] = 0;                                            // dim1_stride hi
  i32x4 z4 = {0, 0, 0, 0};
  i32x8 z8 = {0, 0, 0, 0, 0, 0, 0, 0};
  __builtin_amdgcn_tensor_load_to_lds(g0, g1, z4, z4, z8, 0);
#endif
}

__device__ __forceinline__ void wait_tdm() {
#if defined(HAVE_TDM)
  __builtin_amdgcn_s_wait_tensorcnt(0);
#endif
}

// ---------------------------------------------------------------------------
// f32 -> f16 convert (x)
// ---------------------------------------------------------------------------
__global__ __launch_bounds__(256) void cvt_f32_f16(const float* __restrict__ in,
                                                   _Float16* __restrict__ out,
                                                   int n) {
  int i = blockIdx.x * 256 + threadIdx.x;
  if (i < n) out[i] = (_Float16)in[i];
}

// ---------------------------------------------------------------------------
// Transpose + convert the three weight matrices: T[o][i] = (f16) W[i][o]
// ---------------------------------------------------------------------------
__global__ __launch_bounds__(256) void transpose_cvt3(const float* __restrict__ W0,
                                                      const float* __restrict__ W1,
                                                      const float* __restrict__ W2,
                                                      _Float16* __restrict__ T0,
                                                      _Float16* __restrict__ T1,
                                                      _Float16* __restrict__ T2,
                                                      int D) {
  __shared__ float tile[32][33];
  const float* W = (blockIdx.z == 0) ? W0 : (blockIdx.z == 1) ? W1 : W2;
  _Float16*    T = (blockIdx.z == 0) ? T0 : (blockIdx.z == 1) ? T1 : T2;

  int tx = threadIdx.x & 31;
  int ty = threadIdx.x >> 5;
#pragma unroll
  for (int s = 0; s < 4; ++s) {
    int y = blockIdx.y * 32 + ty + 8 * s;
    tile[ty + 8 * s][tx] = W[(size_t)y * D + blockIdx.x * 32 + tx];
  }
  __syncthreads();
#pragma unroll
  for (int s = 0; s < 4; ++s) {
    int o = blockIdx.x * 32 + ty + 8 * s;
    int i = blockIdx.y * 32 + tx;
    T[(size_t)o * D + i] = (_Float16)tile[tx][ty + 8 * s];
  }
}

// ---------------------------------------------------------------------------
// WMMA f16 GEMM, LDS double-buffered:
//   A tile (256x32) staged by TDM tensor_load_to_lds (wave 0 issues D#),
//   B tile (64x32)  staged by per-lane async global->LDS copies.
//   C[M,N] = A[M,K] * Bt[N,K]^T ; Bt row n = column n of B.
//   STORE: 0 = f32 row-major, 1 = f16 row-major, 2 = f16 transposed (C[n][m])
// Block: 256 threads (8 wave32). Block tile 256(M) x 64(N).
// Wave tile 64x32 = 4x2 WMMA tiles (8 v_wmma per k-step).
// ---------------------------------------------------------------------------
#define LDS_STRIDE 40   // 32 payload halfs + 8 pad (80B rows; TDM pad cfg matches)

template <int STORE>
__global__ __launch_bounds__(256) void gemm_wmma_f16(const _Float16* __restrict__ A,
                                                     const _Float16* __restrict__ Bt,
                                                     void* __restrict__ C,
                                                     int K, int lda, int ldb, int ldc) {
  __shared__ _Float16 As[2][256 * LDS_STRIDE];
  __shared__ _Float16 Bs[2][64 * LDS_STRIDE];

  const int t    = threadIdx.x;
  const int lane = t & 31;
  const int wave = t >> 5;          // 0..7
  const int wm   = wave >> 1;       // 0..3  -> 64-row slab
  const int wn   = wave & 1;        // 0..1  -> 32-col slab
  const int bm0  = blockIdx.x * 256;
  const int bn0  = blockIdx.y * 64;

  const int r  = lane & 15;
  const int hi = lane >> 4;

  v8f acc[4][2] = {};

  // ---- stage one 32-wide k-slab: A (256x32) + Bt (64x32) into LDS buf ----
  auto stage = [&](int k0, int buf) {
#if defined(HAVE_TDM)
    if (wave == 0) {   // one TDM descriptor moves the whole A tile (16KB)
      tdm_load_tile_f16((unsigned)(uintptr_t)&As[buf][0],
                        A + (size_t)bm0 * lda + k0,
                        32u, 256u, (unsigned long long)lda);
    }
#else
#pragma unroll
    for (int q = 0; q < 4; ++q) {
      int c   = t + 256 * q;          // 1024 16B chunks of A tile
      int row = c >> 2;
      int off = (c & 3) * 8;
      copy16_to_lds(&As[buf][row * LDS_STRIDE + off],
                    A + (size_t)(bm0 + row) * lda + k0 + off);
    }
#endif
    {
      int row = t >> 2;               // 256 16B chunks of B tile
      int off = (t & 3) * 8;
      copy16_to_lds(&Bs[buf][row * LDS_STRIDE + off],
                    Bt + (size_t)(bn0 + row) * ldb + k0 + off);
    }
    // pull the k-slab after next toward L2 early
    __builtin_prefetch((const void*)(A + (size_t)(bm0 + (t & 255)) * lda + k0 + 64), 0, 3);
  };

  const int steps = K >> 5;
  stage(0, 0);

  for (int s = 0; s < steps; ++s) {
    const int cur = s & 1;
    wait_async_lds();      // my async copies into buf[cur] complete
    wait_tdm();            // wave 0's tensor DMA into buf[cur] complete
    __syncthreads();       // everyone's staging complete; prev compute done
    if (s + 1 < steps) stage((s + 1) << 5, cur ^ 1);

    // ---- fragments from LDS ----
    v16h a[4], b[2];
#pragma unroll
    for (int i = 0; i < 4; ++i) {
      const _Float16* p = &As[cur][(wm * 64 + 16 * i + r) * LDS_STRIDE + hi * 8];
      v8h lo  = *(const v8h*)p;
      v8h hi8 = *(const v8h*)(p + 16);
      v16h v;
#pragma unroll
      for (int e = 0; e < 8; ++e) { v[e] = lo[e]; v[8 + e] = hi8[e]; }
      a[i] = v;
    }
#pragma unroll
    for (int j = 0; j < 2; ++j) {
      const _Float16* p = &Bs[cur][(wn * 32 + 16 * j + r) * LDS_STRIDE + hi * 8];
      v8h lo  = *(const v8h*)p;
      v8h hi8 = *(const v8h*)(p + 16);
      v16h v;
#pragma unroll
      for (int e = 0; e < 8; ++e) { v[e] = lo[e]; v[8 + e] = hi8[e]; }
      b[j] = v;
    }
    // ---- 8 WMMAs ----
#pragma unroll
    for (int i = 0; i < 4; ++i)
#pragma unroll
      for (int j = 0; j < 2; ++j)
        acc[i][j] = __builtin_amdgcn_wmma_f32_16x16x32_f16(
            false, a[i], false, b[j], (short)0, acc[i][j], false, false);
  }

  // ---- store: C/D layout => element e is row (8*hi + e), col (lane&15) ----
#pragma unroll
  for (int i = 0; i < 4; ++i) {
#pragma unroll
    for (int j = 0; j < 2; ++j) {
      const int col   = bn0 + wn * 32 + 16 * j + r;
      const int mbase = bm0 + wm * 64 + 16 * i + 8 * hi;
      if (STORE == 0) {
        float* Cf = (float*)C;
#pragma unroll
        for (int e = 0; e < 8; ++e)
          Cf[(size_t)(mbase + e) * ldc + col] = acc[i][j][e];
      } else if (STORE == 1) {
        _Float16* Ch = (_Float16*)C;
#pragma unroll
        for (int e = 0; e < 8; ++e)
          Ch[(size_t)(mbase + e) * ldc + col] = (_Float16)acc[i][j][e];
      } else {   // transposed f16: Ct[col][m]
        _Float16* Ct = (_Float16*)C;
        v8h v;
#pragma unroll
        for (int e = 0; e < 8; ++e) v[e] = (_Float16)acc[i][j][e];
        *(v8h*)(Ct + (size_t)col * ldc + mbase) = v;
      }
    }
  }
}

// ---------------------------------------------------------------------------
// Row softmax with scale:  P[r][c] = softmax_c(S[r][c] * scale), f32 -> f16
// ---------------------------------------------------------------------------
__global__ __launch_bounds__(256) void softmax_rows(const float* __restrict__ S,
                                                    _Float16* __restrict__ P,
                                                    int ncols, float scale) {
  const int row = blockIdx.x;
  const float* s = S + (size_t)row * ncols;
  _Float16*    p = P + (size_t)row * ncols;

  __shared__ float red[8];
  float vals[16];

  float m = -3.402823466e+38f;
  int ii = 0;
  for (int c = threadIdx.x; c < ncols; c += 256, ++ii) {
    float v = s[c] * scale;
    vals[ii] = v;
    m = fmaxf(m, v);
  }
#pragma unroll
  for (int o = 16; o > 0; o >>= 1) m = fmaxf(m, __shfl_down(m, o, 32));
  if ((threadIdx.x & 31) == 0) red[threadIdx.x >> 5] = m;
  __syncthreads();
  if (threadIdx.x == 0) {
    float tv = red[0];
#pragma unroll
    for (int i = 1; i < 8; ++i) tv = fmaxf(tv, red[i]);
    red[0] = tv;
  }
  __syncthreads();
  m = red[0];
  __syncthreads();

  float sum = 0.f;
  ii = 0;
  for (int c = threadIdx.x; c < ncols; c += 256, ++ii) {
    float e = expf(vals[ii] - m);
    vals[ii] = e;
    sum += e;
  }
#pragma unroll
  for (int o = 16; o > 0; o >>= 1) sum += __shfl_down(sum, o, 32);
  if ((threadIdx.x & 31) == 0) red[threadIdx.x >> 5] = sum;
  __syncthreads();
  if (threadIdx.x == 0) {
    float tv = 0.f;
#pragma unroll
    for (int i = 0; i < 8; ++i) tv += red[i];
    red[0] = tv;
  }
  __syncthreads();
  const float inv = 1.0f / red[0];

  ii = 0;
  for (int c = threadIdx.x; c < ncols; c += 256, ++ii)
    p[c] = (_Float16)(vals[ii] * inv);
}

// ---------------------------------------------------------------------------
// Launcher
// ---------------------------------------------------------------------------
extern "C" void kernel_launch(void* const* d_in, const int* in_sizes, int n_in,
                              void* d_out, int out_size, void* d_ws, size_t ws_size,
                              hipStream_t stream) {
  const float* x  = (const float*)d_in[0];
  const float* Wq = (const float*)d_in[1];
  const float* Wk = (const float*)d_in[2];
  const float* Wv = (const float*)d_in[3];
  float* out = (float*)d_out;

  const int N = SEQ_N, D = DIM;

  char* ws = (char*)d_ws;
  _Float16* xh  = (_Float16*)ws;  ws += (size_t)N * D * 2;
  _Float16* Wqt = (_Float16*)ws;  ws += (size_t)D * D * 2;
  _Float16* Wkt = (_Float16*)ws;  ws += (size_t)D * D * 2;
  _Float16* Wvt = (_Float16*)ws;  ws += (size_t)D * D * 2;
  _Float16* Q   = (_Float16*)ws;  ws += (size_t)N * D * 2;
  _Float16* Kb  = (_Float16*)ws;  ws += (size_t)N * D * 2;
  _Float16* Vt  = (_Float16*)ws;  ws += (size_t)D * N * 2;   // V transposed [D][N]
  float*    S   = (float*)ws;     ws += (size_t)N * N * 4;
  _Float16* P   = (_Float16*)ws;  ws += (size_t)N * N * 2;

  cvt_f32_f16<<<(N * D + 255) / 256, 256, 0, stream>>>(x, xh, N * D);

  transpose_cvt3<<<dim3(D / 32, D / 32, 3), 256, 0, stream>>>(Wq, Wk, Wv,
                                                              Wqt, Wkt, Wvt, D);

  dim3 gProj(N / 256, D / 64);
  gemm_wmma_f16<1><<<gProj, 256, 0, stream>>>(xh, Wqt, (void*)Q,  D, D, D, D);
  gemm_wmma_f16<1><<<gProj, 256, 0, stream>>>(xh, Wkt, (void*)Kb, D, D, D, D);
  gemm_wmma_f16<2><<<gProj, 256, 0, stream>>>(xh, Wvt, (void*)Vt, D, D, D, N);

  dim3 gScore(N / 256, N / 64);
  gemm_wmma_f16<0><<<gScore, 256, 0, stream>>>(Q, Kb, (void*)S, D, D, D, N);

  const float scale = 1.0f / sqrtf((float)D);
  softmax_rows<<<N, 256, 0, stream>>>(S, P, N, scale);

  dim3 gOut(N / 256, D / 64);
  gemm_wmma_f16<0><<<gOut, 256, 0, stream>>>(P, Vt, (void*)out, N, N, N, D);
}
